// GConv_19164144075285
// MI455X (gfx1250) — matrix operand order repaired and checked by reading, not compiled, not run on hardware
//
#include <hip/hip_runtime.h>
#include <hip/hip_bf16.h>
#include <math.h>

#define D_MODEL 1024
#define KD      64
#define BATCH   8
#define LSEQ    4096
#define NFFT    8192

typedef __attribute__((ext_vector_type(16))) __bf16   v16bf;
typedef __attribute__((ext_vector_type(8)))  __bf16   bfx8;
typedef __attribute__((ext_vector_type(8)))  float    v8f;
typedef __attribute__((ext_vector_type(4)))  int      v4i;
typedef __attribute__((ext_vector_type(4)))  unsigned u32x4;
typedef __attribute__((ext_vector_type(8)))  int      i32x8;

#if __has_include(<hip/amd_detail/amd_gfx1250_TDM.h>)
#warning "PROBE: amd_gfx1250_TDM.h header available"
#endif

#if __has_builtin(__builtin_amdgcn_global_load_async_to_lds_b128) && \
    __has_builtin(__builtin_amdgcn_s_wait_asynccnt)
#define USE_ASYNC_LDS 1
#else
#define USE_ASYNC_LDS 0
#endif

#if __has_builtin(__builtin_amdgcn_tensor_load_to_lds) && \
    __has_builtin(__builtin_amdgcn_s_wait_tensorcnt)
#define USE_TDM 1
#else
#define USE_TDM 0
#endif

// Async copy operand casts. The builtin takes v4i pointers: AS1 source,
// AS3 destination. Flat->LDS uses the ISA 10.2 aperture rule:
// the LDS offset is the low 32 address bits, so truncation of the flat address is exact.
#define GSRC_B128(p) ((__attribute__((address_space(1))) v4i*)(uintptr_t)(p))
#define LDST_B128(p) ((__attribute__((address_space(3))) v4i*)(unsigned)(uintptr_t)(p))

__device__ __forceinline__ float2 cmul(float2 a, float2 b) {
  return make_float2(a.x * b.x - a.y * b.y, a.x * b.y + a.y * b.x);
}

// ---------------------------------------------------------------------------
// Kernel 1: build normalized filter row (piecewise linear upsample of 7 scale
// kernels), forward DIF FFT (output in bit-reversed order), store spectrum.
// One workgroup per d. LDS: 8192 complex = 64 KB (WGP has 320 KB).
// ---------------------------------------------------------------------------
__global__ __launch_bounds__(256)
void k_build_filter_fft(const float* __restrict__ k0, const float* __restrict__ k1,
                        const float* __restrict__ k2, const float* __restrict__ k3,
                        const float* __restrict__ k4, const float* __restrict__ k5,
                        const float* __restrict__ k6,
                        float2* __restrict__ kf /* [D_MODEL][NFFT] */) {
  __shared__ float2 buf[NFFT];
  const int d   = blockIdx.x;
  const int tid = threadIdx.x;

  float ssq = 0.f;
  #pragma unroll
  for (int it = 0; it < LSEQ / 256; ++it) {
    int t = tid + it * 256;
    const float* kp; int start; float s;
    if      (t <   64) { kp = k0; start =    0; s =  1.f; }
    else if (t <  128) { kp = k1; start =   64; s =  1.f; }
    else if (t <  256) { kp = k2; start =  128; s =  2.f; }
    else if (t <  512) { kp = k3; start =  256; s =  4.f; }
    else if (t < 1024) { kp = k4; start =  512; s =  8.f; }
    else if (t < 2048) { kp = k5; start = 1024; s = 16.f; }
    else               { kp = k6; start = 2048; s = 32.f; }
    int   j = t - start;
    float c = ((float)j + 0.5f) / s - 0.5f;
    c = fminf(fmaxf(c, 0.f), (float)(KD - 1));
    int   lo = (int)floorf(c);
    int   hi = min(lo + 1, KD - 1);
    float w  = c - (float)lo;
    const float* kr = kp + d * KD;
    float v = kr[lo] * (1.f - w) + kr[hi] * w;
    buf[t] = make_float2(v, 0.f);
    ssq += v * v;
  }
  // block reduction of sum-of-squares in the zero-pad region
  buf[LSEQ + tid].x = ssq;
  __syncthreads();
  for (int ofs = 128; ofs > 0; ofs >>= 1) {
    if (tid < ofs) buf[LSEQ + tid].x += buf[LSEQ + tid + ofs].x;
    __syncthreads();
  }
  const float inv_norm = rsqrtf(buf[LSEQ].x);
  __syncthreads();
  #pragma unroll
  for (int it = 0; it < LSEQ / 256; ++it) buf[tid + it * 256].x *= inv_norm;
  for (int t = LSEQ + tid; t < NFFT; t += 256) buf[t] = make_float2(0.f, 0.f);
  __syncthreads();

  // forward FFT: radix-2 DIF (Gentleman-Sande), natural in -> bit-reversed out
  for (int len = NFFT; len >= 2; len >>= 1) {
    const int   half = len >> 1;
    const float base = -6.28318530717958647692f / (float)len;
    for (int t = tid; t < NFFT / 2; t += 256) {
      int grp = t / half;
      int j   = t - grp * half;
      int i0  = grp * len + j;
      int i1  = i0 + half;
      float2 a = buf[i0], b = buf[i1];
      float sn, cs; __sincosf(base * (float)j, &sn, &cs);
      buf[i0] = make_float2(a.x + b.x, a.y + b.y);
      buf[i1] = cmul(make_float2(a.x - b.x, a.y - b.y), make_float2(cs, sn));
    }
    __syncthreads();
  }
  float2* dst = kf + (size_t)d * NFFT;
  for (int t = tid; t < NFFT; t += 256) dst[t] = buf[t];
}

// ---------------------------------------------------------------------------
// Kernel 2: per (b,d) row — forward DIF FFT of u (zero-padded), pointwise
// multiply with the stored bit-reversed k-spectrum, inverse DIT FFT
// (bit-reversed in -> natural out), +u*D, exact GELU, cast to bf16.
// u row is staged global->LDS with the async copy path (ASYNCcnt) into the
// region that later becomes FFT zero-padding, then expanded to complex.
// ---------------------------------------------------------------------------
__global__ __launch_bounds__(256)
void k_fftconv(const float* __restrict__ u, const float2* __restrict__ kf,
               const float* __restrict__ Dv, __bf16* __restrict__ ybf) {
  __shared__ float2 buf[NFFT];
  const int row = blockIdx.x;            // b * D_MODEL + d
  const int d   = row & (D_MODEL - 1);
  const int tid = threadIdx.x;
  const float* up = u + (size_t)row * LSEQ;

#if USE_ASYNC_LDS
  {
    float* raw = (float*)&buf[NFFT - LSEQ / 2];   // 16 KB tail scratch
    #pragma unroll
    for (int p = 0; p < 4; ++p) {
      const int c = tid + p * 256;                // 16-byte chunk index
      __builtin_amdgcn_global_load_async_to_lds_b128(GSRC_B128(up + c * 4),
                                                     LDST_B128(raw + c * 4), 0, 0);
    }
    __builtin_amdgcn_s_wait_asynccnt(0);
    __syncthreads();
    for (int t = tid; t < LSEQ; t += 256) buf[t] = make_float2(raw[t], 0.f);
    __syncthreads();
  }
#else
  for (int t = tid; t < LSEQ; t += 256) buf[t] = make_float2(up[t], 0.f);
#endif
  for (int t = LSEQ + tid; t < NFFT; t += 256) buf[t] = make_float2(0.f, 0.f);
  __syncthreads();

  // forward DIF
  for (int len = NFFT; len >= 2; len >>= 1) {
    const int   half = len >> 1;
    const float base = -6.28318530717958647692f / (float)len;
    for (int t = tid; t < NFFT / 2; t += 256) {
      int grp = t / half;
      int j   = t - grp * half;
      int i0  = grp * len + j;
      int i1  = i0 + half;
      float2 a = buf[i0], b = buf[i1];
      float sn, cs; __sincosf(base * (float)j, &sn, &cs);
      buf[i0] = make_float2(a.x + b.x, a.y + b.y);
      buf[i1] = cmul(make_float2(a.x - b.x, a.y - b.y), make_float2(cs, sn));
    }
    __syncthreads();
  }

  // pointwise multiply (both spectra in the identical bit-reversed order)
  const float2* kr = kf + (size_t)d * NFFT;
  for (int t = tid; t < NFFT; t += 256) buf[t] = cmul(buf[t], kr[t]);
  __syncthreads();

  // inverse DIT (Cooley-Tukey), bit-reversed in -> natural out
  for (int len = 2; len <= NFFT; len <<= 1) {
    const int   half = len >> 1;
    const float base = 6.28318530717958647692f / (float)len;
    for (int t = tid; t < NFFT / 2; t += 256) {
      int grp = t / half;
      int j   = t - grp * half;
      int i0  = grp * len + j;
      int i1  = i0 + half;
      float sn, cs; __sincosf(base * (float)j, &sn, &cs);
      float2 bw = cmul(buf[i1], make_float2(cs, sn));
      float2 a  = buf[i0];
      buf[i0] = make_float2(a.x + bw.x, a.y + bw.y);
      buf[i1] = make_float2(a.x - bw.x, a.y - bw.y);
    }
    __syncthreads();
  }

  const float dval = Dv[d];
  const float invN = 1.f / (float)NFFT;
  __bf16* yp = ybf + (size_t)row * LSEQ;
  for (int t = tid; t < LSEQ; t += 256) {
    float y = buf[t].x * invN + up[t] * dval;
    float g = 0.5f * y * (1.f + erff(y * 0.70710678118654752f));  // exact GELU
    yp[t] = (__bf16)g;
  }
}

// ---------------------------------------------------------------------------
// Kernel 3: W (fp32, row-major [v][u]) -> bf16
// ---------------------------------------------------------------------------
__global__ __launch_bounds__(256)
void k_w2bf(const float* __restrict__ W, __bf16* __restrict__ Wb, int n) {
  int i = blockIdx.x * 256 + threadIdx.x;
  if (i < n) Wb[i] = (__bf16)W[i];
}

// ---------------------------------------------------------------------------
// Kernel 4: out[b,v,l] = sum_u W[v,u] * y[b,u,l] + bias[v]
// bf16 WMMA GEMM, per-batch M=1024, N=4096, K=1024.
// Block tile 128x128, BK=32; 8 waves (wave32), each wave owns a 64x32 tile
// = 4x2 v_wmma_f32_16x16x32_bf16 accumulators.
//  - A tile (128x32 of W) staged by the Tensor Data Mover: one 2D descriptor
//    per K-step, LDS padding feature producing the 80 B row pitch directly
//    (64 B data + 16 B pad). TENSORcnt + workgroup barrier publish it.
//  - B tile staged pre-swizzled to [n_tile][n][k] so every fragment is two
//    aligned ds_load_b128 per lane (no scalar u16 gathers).
// ---------------------------------------------------------------------------
#define BM   128
#define BN   128
#define BK   32
#define KPAD 40   // row stride 80 B: keeps every 16 B sub-chunk aligned

__global__ __launch_bounds__(256)
void k_gemm_bias(const __bf16* __restrict__ A  /* Wb [1024][1024] */,
                 const __bf16* __restrict__ Bm /* ybf [B][1024][4096] */,
                 const float* __restrict__ bias,
                 float* __restrict__ out) {
  __shared__ __bf16 sA[BM][KPAD];       // [m][k], 80 B pitch
  __shared__ __bf16 sBs[8][16][KPAD];   // [n_tile][n][k], k contiguous
  const int tid  = threadIdx.x;
  const int lane = tid & 31;
  const int wave = tid >> 5;
  const int wm   = wave >> 2;   // 0..1 -> 64-row band
  const int wn   = wave & 3;    // 0..3 -> 32-col band
  const int bz   = blockIdx.z;
  const int m0   = blockIdx.y * BM;
  const int n0   = blockIdx.x * BN;
  const __bf16* Bbase = Bm + (size_t)bz * D_MODEL * LSEQ;

  v8f zero = {};
  v8f acc[4][2];
  #pragma unroll
  for (int i = 0; i < 4; ++i)
    #pragma unroll
    for (int j = 0; j < 2; ++j) acc[i][j] = zero;

  const int arow  = tid >> 2;         // 0..63 (+64 second pass)
  const int acol  = (tid & 3) * 8;    // 8-wide chunks over K
  const int brow  = tid >> 4;         // 0..15 (+16 second pass) -> k
  const int bcol  = (tid & 15) * 8;   // column chunk over N
  const int ntile = bcol >> 4;        // 16-col tile index (0..7)
  const int nloc  = bcol & 15;        // 0 or 8 within tile

  for (int k0 = 0; k0 < D_MODEL; k0 += BK) {
    __syncthreads();

    // ---- stage A tile (128 rows x 32 k) ----
#if USE_TDM
    if (tid < 32) {  // one TDM descriptor per workgroup (EXEC-independent, wave-level)
      const unsigned long long ga = (unsigned long long)(uintptr_t)
                                    (A + (size_t)m0 * D_MODEL + k0);
      const unsigned lb = (unsigned)(uintptr_t)&sA[0][0];  // LDS byte offset
      // D# group 0: count=1 | lds_addr | global_addr[56:0] | type=2
      u32x4 g0 = { 1u,
                   lb,
                   (unsigned)ga,
                   (unsigned)((ga >> 32) & 0x1FFFFFFu) | (2u << 30) };
      // D# group 1: data_size=2B, pad_enable, pad_interval=16 DW (64 B),
      // pad_amount=4 DW (16 B); tensor 1024x1024, tile 32x128, stride 1024.
      i32x8 g1 = { (int)((1u << 16) | (1u << 20) | (3u << 22) | (3u << 25)),
                   (int)(1024u << 16),   // tensor_dim0[15:0]
                   (int)(1024u << 16),   // tensor_dim0[31:16]=0 | tensor_dim1[15:0]
                   (int)(32u << 16),     // tensor_dim1[31:16]=0 | tile_dim0=32
                   128,                  // tile_dim1=128 | tile_dim2=0
                   1024,                 // tensor_dim0_stride[31:0]
                   0, 0 };
      v4i   gz4 = {};                    // groups 2/3 unused (2D tensor)
      i32x8 gz8 = {};                    // extra group (clang-23 6-arg form)
      __builtin_amdgcn_tensor_load_to_lds(g0, g1, gz4, gz4, gz8, 0);
      __builtin_amdgcn_s_wait_tensorcnt(0);
    }
#else
    #pragma unroll
    for (int p = 0; p < 2; ++p) {
      const int r = arow + p * 64;
      const __bf16* src = A + (size_t)(m0 + r) * D_MODEL + k0 + acol;
#if USE_ASYNC_LDS
      __builtin_amdgcn_global_load_async_to_lds_b128(GSRC_B128(src),
                                                     LDST_B128(&sA[r][acol]), 0, 0);
#else
      *(bfx8*)&sA[r][acol] = *(const bfx8*)src;
#endif
    }
#endif
    // ---- stage B tile (32 x 128), swizzled to [n_tile][n][k] ----
    #pragma unroll
    for (int p = 0; p < 2; ++p) {
      const int k = brow + p * 16;
      bfx8 v = *(const bfx8*)(Bbase + (size_t)(k0 + k) * LSEQ + n0 + bcol);
      #pragma unroll
      for (int e = 0; e < 8; ++e) sBs[ntile][nloc + e][k] = v[e];
    }
    if (k0 + BK < D_MODEL) {   // warm caches for next tiles
      __builtin_prefetch(A + (size_t)(m0 + arow) * D_MODEL + k0 + BK + acol, 0, 3);
      __builtin_prefetch(Bbase + (size_t)(k0 + BK + brow) * LSEQ + n0 + bcol, 0, 3);
    }
#if !USE_TDM && USE_ASYNC_LDS
    __builtin_amdgcn_s_wait_asynccnt(0);
#endif
    __syncthreads();

    // ---- fragments: ISA 16-bit A 16x32 / B 32x16 lane striping ----
    // A: lane L<16 -> row M=L, K chunks {0..7,16..23}; L>=16 -> +8
    const int kA = (lane >> 4) ? 8 : 0;
    v16bf af[4];
    #pragma unroll
    for (int mt = 0; mt < 4; ++mt) {
      const int m = wm * 64 + mt * 16 + (lane & 15);
      bfx8 lo = *(const bfx8*)&sA[m][kA];
      bfx8 hi = *(const bfx8*)&sA[m][16 + kA];
      af[mt] = __builtin_shufflevector(lo, hi, 0, 1, 2, 3, 4, 5, 6, 7,
                                               8, 9, 10, 11, 12, 13, 14, 15);
    }
    // B: lane L<16 -> col N=L, K chunks {0..7,16..23}; L>=16 -> col L-16, +8
    const int kB = (lane >> 4) ? 8 : 0;
    v16bf bfrag[2];
    #pragma unroll
    for (int nt = 0; nt < 2; ++nt) {
      const int tt = wn * 2 + nt;
      const int n  = lane & 15;
      bfx8 lo = *(const bfx8*)&sBs[tt][n][kB];
      bfx8 hi = *(const bfx8*)&sBs[tt][n][kB + 16];
      bfrag[nt] = __builtin_shufflevector(lo, hi, 0, 1, 2, 3, 4, 5, 6, 7,
                                                  8, 9, 10, 11, 12, 13, 14, 15);
    }

    #pragma unroll
    for (int mt = 0; mt < 4; ++mt)
      #pragma unroll
      for (int nt = 0; nt < 2; ++nt)
        acc[mt][nt] = __builtin_amdgcn_wmma_f32_16x16x32_bf16(
            false, af[mt], false, bfrag[nt], (short)0, acc[mt][nt], false, false);
  }

  // epilogue: C 16x16 layout — VGPR r: lanes 0-15 -> M=r, lanes 16-31 -> M=8+r
  const int rofs = (lane >> 4) ? 8 : 0;
  #pragma unroll
  for (int mt = 0; mt < 4; ++mt)
    #pragma unroll
    for (int nt = 0; nt < 2; ++nt)
      #pragma unroll
      for (int r = 0; r < 8; ++r) {
        const int row = m0 + wm * 64 + mt * 16 + r + rofs;
        const int col = n0 + wn * 32 + nt * 16 + (lane & 15);
        out[((size_t)bz * D_MODEL + row) * LSEQ + col] = acc[mt][nt][r] + bias[row];
      }
}

// ---------------------------------------------------------------------------
extern "C" void kernel_launch(void* const* d_in, const int* in_sizes, int n_in,
                              void* d_out, int out_size, void* d_ws, size_t ws_size,
                              hipStream_t stream) {
  (void)in_sizes; (void)n_in; (void)out_size; (void)ws_size;
  const float* u    = (const float*)d_in[0];
  const float* k0   = (const float*)d_in[1];
  const float* k1   = (const float*)d_in[2];
  const float* k2   = (const float*)d_in[3];
  const float* k3   = (const float*)d_in[4];
  const float* k4   = (const float*)d_in[5];
  const float* k5   = (const float*)d_in[6];
  const float* k6   = (const float*)d_in[7];
  const float* Dv   = (const float*)d_in[8];
  const float* W    = (const float*)d_in[9];
  const float* bias = (const float*)d_in[10];
  float* out = (float*)d_out;

  // workspace layout: k-spectrum (64 MB) | y_bf16 (64 MB) | W_bf16 (2 MB)
  char*   ws  = (char*)d_ws;
  float2* kf  = (float2*)ws;
  __bf16* ybf = (__bf16*)(ws + (size_t)D_MODEL * NFFT * sizeof(float2));
  __bf16* wbf = (__bf16*)((char*)ybf + (size_t)BATCH * D_MODEL * LSEQ * sizeof(__bf16));

  k_build_filter_fft<<<D_MODEL, 256, 0, stream>>>(k0, k1, k2, k3, k4, k5, k6, kf);
  k_w2bf<<<(D_MODEL * D_MODEL + 255) / 256, 256, 0, stream>>>(W, wbf, D_MODEL * D_MODEL);
  k_fftconv<<<BATCH * D_MODEL, 256, 0, stream>>>(u, kf, Dv, ybf);
  k_gemm_bias<<<dim3(LSEQ / BN, D_MODEL / BM, BATCH), 256, 0, stream>>>(wbf, ybf, bias, out);
}
